// MultiHeadAttention_28441273434366
// MI455X (gfx1250) — compile-verified, matrix operand order checked by reading, not run on hardware
//
#include <hip/hip_runtime.h>
#include <hip/hip_bf16.h>

// MHA forward for MI455X (gfx1250, wave32, WMMA).
// Pipeline: f32->f16 convert (activations & weights, one pass each)
//           -> 3x pure-f16 projection GEMM (V stored head-transposed)
//           -> flash attention (f16 WMMA, f32 accum, online softmax)
//           -> output projection (f16 in -> f32 out).
// Workspace: 5 x (B*L*D) f16 + 1 x (D*D) f16 = 42 MB.

typedef __attribute__((ext_vector_type(16))) _Float16 v16h;
typedef __attribute__((ext_vector_type(8)))  _Float16 v8h;
typedef __attribute__((ext_vector_type(8)))  float    v8f;

constexpr int B  = 2;
constexpr int L  = 2048;
constexpr int D  = 1024;
constexpr int H  = 16;
constexpr int DH = 64;
constexpr int M  = B * L;              // 4096 token rows
constexpr float NEG_BIG = -1e30f;

__device__ __forceinline__ v16h ld16h(const _Float16* __restrict__ p) {
  return *reinterpret_cast<const v16h*>(p);
}
__device__ __forceinline__ v8f wmma_f16(v16h a, v16h b, v8f c) {
  // D(f32 16x16) = A(16x32 f16) * B(32x16 f16) + C
  return __builtin_amdgcn_wmma_f32_16x16x32_f16(false, a, false, b,
                                                (short)0, c, false, false);
}

// ---------------------------------------------------------------------------
// Elementwise f32 -> f16, 8 elements/thread (b128 load -> b128 store).
// ---------------------------------------------------------------------------
__global__ __launch_bounds__(256) void cvt_f16_kernel(
    const float* __restrict__ in, _Float16* __restrict__ out, int n8)
{
  const int i = blockIdx.x * blockDim.x + threadIdx.x;
  if (i >= n8) return;
  const float4* p = reinterpret_cast<const float4*>(in) + (size_t)i * 2;
  const float4 a = p[0], b = p[1];
  v8h r;
  r[0] = (_Float16)a.x; r[1] = (_Float16)a.y;
  r[2] = (_Float16)a.z; r[3] = (_Float16)a.w;
  r[4] = (_Float16)b.x; r[5] = (_Float16)b.y;
  r[6] = (_Float16)b.z; r[7] = (_Float16)b.w;
  *reinterpret_cast<v8h*>(out + (size_t)i * 8) = r;
}

// ---------------------------------------------------------------------------
// GEMM: out[m,n] = sum_k A[m,k] * W[n,k] + bias[n]   (torch Linear x@W^T+b)
// A, W already f16. One wave computes a 16x64 tile (4 accumulators); the A
// fragment is reused across 4 WMMAs per 32-wide K step.
// OUT_MODE: 0 = f16 row-major [M,D], 1 = f16 per-head transposed V layout
//           [(b*H+h)*DH+d][j] (N-tile 64 == DH so h == nt), 2 = f32 [M,D].
// ---------------------------------------------------------------------------
template <int OUT_MODE>
__global__ __launch_bounds__(256) void gemm_kernel(
    const _Float16* __restrict__ A, const _Float16* __restrict__ W,
    const float* __restrict__ bias, void* __restrict__ outp)
{
  constexpr int NT = D / 64;                       // 16 column super-tiles
  const int lane = threadIdx.x & 31;
  const int wid  = (blockIdx.x * blockDim.x + threadIdx.x) >> 5;
  if (wid >= (M / 16) * NT) return;                // wave-uniform guard
  const int mt = wid / NT, nt = wid % NT;
  const int g  = lane >> 4;                        // K-half selector
  const int hn = lane & 15;
  const int row = mt * 16 + hn;                    // A row for this lane

  const _Float16* wr0 = &W[(size_t)(nt * 64 +  0 + hn) * D];
  const _Float16* wr1 = &W[(size_t)(nt * 64 + 16 + hn) * D];
  const _Float16* wr2 = &W[(size_t)(nt * 64 + 32 + hn) * D];
  const _Float16* wr3 = &W[(size_t)(nt * 64 + 48 + hn) * D];
  const _Float16* ar  = &A[(size_t)row * D];

  v8f acc0 = {}, acc1 = {}, acc2 = {}, acc3 = {};
#pragma unroll 2
  for (int kk = 0; kk < D; kk += 32) {
    const int k0 = kk + g * 16;
    const v16h a = ld16h(ar + k0);                 // reused 4x
    acc0 = wmma_f16(a, ld16h(wr0 + k0), acc0);
    acc1 = wmma_f16(a, ld16h(wr1 + k0), acc1);
    acc2 = wmma_f16(a, ld16h(wr2 + k0), acc2);
    acc3 = wmma_f16(a, ld16h(wr3 + k0), acc3);
  }

#pragma unroll
  for (int c = 0; c < 4; ++c) {
    const v8f acc = (c == 0) ? acc0 : (c == 1) ? acc1 : (c == 2) ? acc2 : acc3;
    const int col = nt * 64 + c * 16 + hn;
    const float bv = bias[col];
#pragma unroll
    for (int r = 0; r < 8; ++r) {
      const int m = mt * 16 + r + 8 * g;           // output row (C layout)
      const float val = acc[r] + bv;
      if constexpr (OUT_MODE == 2) {
        ((float*)outp)[(size_t)m * D + col] = val;
      } else if constexpr (OUT_MODE == 0) {
        ((_Float16*)outp)[(size_t)m * D + col] = (_Float16)val;
      } else {                                     // head-transposed V: h == nt
        const int d = c * 16 + hn;
        const int bb = m / L, j = m % L;
        ((_Float16*)outp)[((size_t)(bb * H + nt) * DH + d) * L + j] = (_Float16)val;
      }
    }
  }
}

// ---------------------------------------------------------------------------
// Flash attention: one wave per (b, h, 16-query tile). Keys in blocks of 32.
// S = (Q/sqrt(DH)) @ K^T  (2 C-tiles, K-dim DH=64 -> 2 WMMA each)
// online softmax -> P (f16, staged through per-wave LDS tile to A-fragment)
// O(16x64) += P(16x32) @ V(32x64)  (4 WMMA, one per 16-wide d chunk).
// Causal mask implemented analytically (attn_mask input is the causal triu);
// key_padding_mask applied per key column.
// ---------------------------------------------------------------------------
__global__ __launch_bounds__(256) void flash_kernel(
    const _Float16* __restrict__ Qh, const _Float16* __restrict__ Kh,
    const _Float16* __restrict__ Vt, const unsigned char* __restrict__ kpad,
    _Float16* __restrict__ Out)
{
  __shared__ _Float16 lds_p[8][16][32];            // 1 KB per wave
  const int lane = threadIdx.x & 31;
  const int wv   = threadIdx.x >> 5;
  const int wid  = blockIdx.x * 8 + wv;
  constexpr int QT = L / 16;                       // 128 query tiles
  const int b  = wid / (H * QT);
  const int h  = (wid / QT) % H;
  const int qt = wid % QT;
  const int qbase = qt * 16;
  const int g = lane >> 4, n = lane & 15;

  // Q A-fragments, pre-scaled by 1/sqrt(DH) = 0.125
  const _Float16* qrow = &Qh[((size_t)(b * L + qbase + n)) * D + h * DH];
  v16h aQ0, aQ1;
#pragma unroll
  for (int i = 0; i < 16; ++i) {
    aQ0[i] = (_Float16)(0.125f * (float)qrow[g * 16 + i]);
    aQ1[i] = (_Float16)(0.125f * (float)qrow[32 + g * 16 + i]);
  }

  v8f O0 = {}, O1 = {}, O2 = {}, O3 = {};
  float mrun[8], lrun[8];
#pragma unroll
  for (int r = 0; r < 8; ++r) { mrun[r] = NEG_BIG; lrun[r] = 0.f; }

  const int jend = qbase + 16;                     // causal: skip j > q_max
  for (int jb = 0; jb < jend; jb += 32) {
    // prefetch next key block's K rows into cache (global_prefetch_b8)
    if (jb + 32 < jend) {
      __builtin_prefetch(&Kh[((size_t)(b * L + jb + 32 + n)) * D + h * DH], 0, 3);
      __builtin_prefetch(&Kh[((size_t)(b * L + jb + 48 + n)) * D + h * DH], 0, 3);
    }
    // ---- S = Qs @ K^T : two 16x16 C tiles over 32 keys
    const _Float16* k0p = &Kh[((size_t)(b * L + jb + n)) * D + h * DH];
    const _Float16* k1p = &Kh[((size_t)(b * L + jb + 16 + n)) * D + h * DH];
    v8f s0 = {}, s1 = {};
    s0 = wmma_f16(aQ0, ld16h(k0p + g * 16),      s0);
    s0 = wmma_f16(aQ1, ld16h(k0p + 32 + g * 16), s0);
    s1 = wmma_f16(aQ0, ld16h(k1p + g * 16),      s1);
    s1 = wmma_f16(aQ1, ld16h(k1p + 32 + g * 16), s1);

    // ---- masks + online softmax (row m = r + 8*g, key col = lane's n)
    const int j0 = jb + n, j1 = jb + 16 + n;
    const bool pad0 = kpad[b * L + j0] != 0;
    const bool pad1 = kpad[b * L + j1] != 0;
    float sc[8], e0v[8], e1v[8];
#pragma unroll
    for (int r = 0; r < 8; ++r) {
      const int q = qbase + r + 8 * g;
      float x0 = s0[r], x1 = s1[r];
      if (pad0 || (j0 > q)) x0 = NEG_BIG;          // per-lane select, no EXEC div.
      if (pad1 || (j1 > q)) x1 = NEG_BIG;
      float rm = fmaxf(x0, x1);
      rm = fmaxf(rm, __shfl_xor(rm, 1, 16));
      rm = fmaxf(rm, __shfl_xor(rm, 2, 16));
      rm = fmaxf(rm, __shfl_xor(rm, 4, 16));
      rm = fmaxf(rm, __shfl_xor(rm, 8, 16));
      const float newm  = fmaxf(mrun[r], rm);
      const float scale = __expf(mrun[r] - newm);
      const float e0 = __expf(x0 - newm);
      const float e1 = __expf(x1 - newm);
      float rs = e0 + e1;
      rs += __shfl_xor(rs, 1, 16);
      rs += __shfl_xor(rs, 2, 16);
      rs += __shfl_xor(rs, 4, 16);
      rs += __shfl_xor(rs, 8, 16);
      lrun[r] = lrun[r] * scale + rs;
      mrun[r] = newm;
      sc[r] = scale; e0v[r] = e0; e1v[r] = e1;
    }
    // rescale O, stage P (C layout -> LDS row-major [m][k])
#pragma unroll
    for (int r = 0; r < 8; ++r) {
      O0[r] *= sc[r]; O1[r] *= sc[r]; O2[r] *= sc[r]; O3[r] *= sc[r];
      const int m = r + 8 * g;
      lds_p[wv][m][n]      = (_Float16)e0v[r];
      lds_p[wv][m][n + 16] = (_Float16)e1v[r];
    }
    asm volatile("s_wait_dscnt 0" ::: "memory");   // wave-private DS RAW

    // ---- O += P @ V : P A-fragment from LDS, V B-fragments from Vt (contig j)
    const v16h aP = *reinterpret_cast<const v16h*>(&lds_p[wv][n][g * 16]);
    const size_t vb = ((size_t)(b * H + h) * DH + n) * L + jb + g * 16;
    O0 = wmma_f16(aP, ld16h(&Vt[vb]),                  O0);
    O1 = wmma_f16(aP, ld16h(&Vt[vb + (size_t)16 * L]), O1);
    O2 = wmma_f16(aP, ld16h(&Vt[vb + (size_t)32 * L]), O2);
    O3 = wmma_f16(aP, ld16h(&Vt[vb + (size_t)48 * L]), O3);
  }

  // ---- normalize and store f16 attention output [b, q, h*DH + d]
#pragma unroll
  for (int r = 0; r < 8; ++r) {
    const float inv = lrun[r] > 0.f ? 1.0f / lrun[r] : 0.f;
    const int q = qbase + r + 8 * g;
    _Float16* op = &Out[((size_t)(b * L + q)) * D + h * DH];
    op[n]      = (_Float16)(O0[r] * inv);
    op[16 + n] = (_Float16)(O1[r] * inv);
    op[32 + n] = (_Float16)(O2[r] * inv);
    op[48 + n] = (_Float16)(O3[r] * inv);
  }
}

// ---------------------------------------------------------------------------
extern "C" void kernel_launch(void* const* d_in, const int* in_sizes, int n_in,
                              void* d_out, int out_size, void* d_ws, size_t ws_size,
                              hipStream_t stream) {
  (void)in_sizes; (void)n_in; (void)out_size; (void)ws_size;
  const float* q  = (const float*)d_in[0];
  const float* k  = (const float*)d_in[1];
  const float* v  = (const float*)d_in[2];
  // d_in[3] = attn_mask (causal triu) -- applied analytically in flash_kernel
  const unsigned char* kpad = (const unsigned char*)d_in[4];
  const float* Wq = (const float*)d_in[5];  const float* bq = (const float*)d_in[6];
  const float* Wk = (const float*)d_in[7];  const float* bk = (const float*)d_in[8];
  const float* Wv = (const float*)d_in[9];  const float* bv = (const float*)d_in[10];
  const float* Wo = (const float*)d_in[11]; const float* bo = (const float*)d_in[12];
  float* out = (float*)d_out;

  const size_t act = (size_t)M * D;                // 4 Mi f16 per region
  const size_t wgt = (size_t)D * D;                // 1 Mi f16
  _Float16* Qh = (_Float16*)d_ws;                  // needs ~42 MB workspace
  _Float16* Kh = Qh + act;
  _Float16* Vt = Kh + act;
  _Float16* At = Vt + act;
  _Float16* Xh = At + act;                         // converted activation (reused)
  _Float16* Wh = Xh + act;                         // converted weight (reused)

  const int cvtA_blk = (int)(act / 8 / 256);       // 2048
  const int cvtW_blk = (int)(wgt / 8 / 256);       // 512
  const int gemm_blk = ((M / 16) * (D / 64)) / 8;  // 4096 waves -> 512 blocks
  const int fa_blk   = (B * H * (L / 16)) / 8;     // 4096 waves -> 512 blocks

  // Q projection
  cvt_f16_kernel<<<cvtA_blk, 256, 0, stream>>>(q, Xh, (int)(act / 8));
  cvt_f16_kernel<<<cvtW_blk, 256, 0, stream>>>(Wq, Wh, (int)(wgt / 8));
  gemm_kernel<0><<<gemm_blk, 256, 0, stream>>>(Xh, Wh, bq, Qh);
  // K projection
  cvt_f16_kernel<<<cvtA_blk, 256, 0, stream>>>(k, Xh, (int)(act / 8));
  cvt_f16_kernel<<<cvtW_blk, 256, 0, stream>>>(Wk, Wh, (int)(wgt / 8));
  gemm_kernel<0><<<gemm_blk, 256, 0, stream>>>(Xh, Wh, bk, Kh);
  // V projection (head-transposed output)
  cvt_f16_kernel<<<cvtA_blk, 256, 0, stream>>>(v, Xh, (int)(act / 8));
  cvt_f16_kernel<<<cvtW_blk, 256, 0, stream>>>(Wv, Wh, (int)(wgt / 8));
  gemm_kernel<1><<<gemm_blk, 256, 0, stream>>>(Xh, Wh, bv, Vt);
  // Attention
  flash_kernel<<<fa_blk, 256, 0, stream>>>(Qh, Kh, Vt, kpad, At);
  // Output projection (f32 out)
  cvt_f16_kernel<<<cvtW_blk, 256, 0, stream>>>(Wo, Wh, (int)(wgt / 8));
  gemm_kernel<2><<<gemm_blk, 256, 0, stream>>>(At, Wh, bo, out);
}
